// StepRnnSparseDynamic_40776419508801
// MI455X (gfx1250) — matrix operand order repaired and checked by reading, not compiled, not run on hardware
//
#include <hip/hip_runtime.h>

// Problem dims (fixed by the reference)
#define B_  64
#define T_  512
#define A_  128
#define H_  512
#define K_  (A_ + H_)   // 640 : concatenated [x | h] contraction dim
#define G4_ (4 * H_)    // 2048 gate columns (i,f,g,o)

typedef __attribute__((ext_vector_type(16))) __bf16        v16bf;
typedef __attribute__((ext_vector_type(8)))  float         v8f;
typedef __attribute__((ext_vector_type(4)))  unsigned int  u32x4;

union FragBF { v16bf bf; u32x4 q[2]; };   // 16 bf16 = 32B = two 16B chunks

// ---- workspace layout (bytes) ----
#define WCAT_OFF 0u                          // [4H][K] bf16 : 2,621,440
#define WOUT_OFF 2621440u                    // [A][H]  bf16 :   131,072
#define BIAS_OFF 2752512u                    // [4H]    f32  :     8,192
#define MBF_OFF  2760704u                    // [B][H][H] bf16: 33,554,432
#define HALL_OFF 36315136u                   // [B][T][H] bf16: 33,554,432
// total ~66.6 MiB

__device__ __forceinline__ unsigned short f2bf(float f) {
    unsigned int u = __float_as_uint(f);
    return (unsigned short)((u + 0x7fffu + ((u >> 16) & 1u)) >> 16);  // RNE
}
__device__ __forceinline__ float bf2f(unsigned short s) {
    return __uint_as_float(((unsigned int)s) << 16);
}
__device__ __forceinline__ float sigm(float x) {
    return __builtin_amdgcn_rcpf(1.f + __expf(-x));
}
__device__ __forceinline__ float tanh_f(float x) {
    float e = __expf(-2.f * x);
    return (1.f - e) * __builtin_amdgcn_rcpf(1.f + e);
}

// ---------------------------------------------------------------------------
// Kernel 0: one-time weight prep: bias = b_ih + b_hh, bf16 copies of
// Wcat = [W_ih | W_hh] (row-major [4H][640]), W_out, and M.
// ---------------------------------------------------------------------------
__global__ void prep_kernel(const float* __restrict__ w_ih,
                            const float* __restrict__ w_hh,
                            const float* __restrict__ b_ih,
                            const float* __restrict__ b_hh,
                            const float* __restrict__ M,
                            const float* __restrict__ w_out,
                            unsigned short* __restrict__ wcat,
                            unsigned short* __restrict__ wout,
                            float* __restrict__ bias,
                            unsigned short* __restrict__ mbf) {
    const int stride = gridDim.x * blockDim.x;
    const int tid = blockIdx.x * blockDim.x + threadIdx.x;
    for (int i = tid; i < G4_; i += stride) bias[i] = b_ih[i] + b_hh[i];
    for (int i = tid; i < G4_ * K_; i += stride) {
        int c = i / K_, k = i - c * K_;
        float v = (k < A_) ? w_ih[c * A_ + k] : w_hh[c * H_ + (k - A_)];
        wcat[i] = f2bf(v);
    }
    for (int i = tid; i < A_ * H_; i += stride) wout[i] = f2bf(w_out[i]);
    for (size_t i = tid; i < (size_t)B_ * H_ * H_; i += stride) mbf[i] = f2bf(M[i]);
}

// ---------------------------------------------------------------------------
// Kernel 1: serial LSTM recurrence. 4 workgroups (one per 16-sample batch
// tile) x 1024 threads (32 waves). Wave w owns hidden slice j0 = 16*w for all
// four gates; c-state lives in WMMA accumulator layout in registers for all
// 512 steps; h is exchanged through LDS. Per K-iteration: all 4 B fragments
// are loaded into distinct registers first (single load clause) so the four
// v_wmma issues overlap with the tail of the L2 loads.
// ---------------------------------------------------------------------------
__global__ void __launch_bounds__(1024)
lstm_recurrent_kernel(const float* __restrict__ inputs,
                      const unsigned short* __restrict__ wcat,
                      const float* __restrict__ bias,
                      const float* __restrict__ h0,
                      const float* __restrict__ c0,
                      unsigned short* __restrict__ h_all) {
    __shared__ unsigned short xh[16][K_];    // [sample][x(128) | h(512)] bf16, 20KB

    const int bt   = blockIdx.x;             // batch tile (16 samples)
    const int tid  = threadIdx.x;
    const int wave = tid >> 5;
    const int lane = tid & 31;
    const int n    = lane & 15;              // WMMA column-in-tile / sample row
    const int kh   = lane >> 4;              // K-half selector
    const int j0   = wave * 16;              // hidden slice base

    // ---- init LDS with x_0 and h0 ----
    for (int e = tid; e < 16 * A_; e += 1024) {
        int m = e >> 7, a = e & (A_ - 1);
        xh[m][a] = f2bf(inputs[((size_t)(bt * 16 + m) * T_) * A_ + a]);
    }
    for (int e = tid; e < 16 * H_; e += 1024) {
        int m = e >> 9, j = e & (H_ - 1);
        xh[m][A_ + j] = f2bf(h0[(size_t)(bt * 16 + m) * H_ + j]);
    }
    // ---- c state in registers, laid out like the WMMA C/D matrix:
    //      element r of the v8f = (sample m = r + 8*kh, hidden j = j0 + n)
    v8f creg;
#pragma unroll
    for (int r = 0; r < 8; ++r)
        creg[r] = c0[(size_t)(bt * 16 + r + 8 * kh) * H_ + j0 + n];
    float bval[4];
#pragma unroll
    for (int g = 0; g < 4; ++g) bval[g] = bias[g * H_ + j0 + n];

    __syncthreads();

#pragma unroll 1
    for (int t = 0; t < T_; ++t) {
        v8f acc[4];
#pragma unroll
        for (int g = 0; g < 4; ++g) {
#pragma unroll
            for (int r = 0; r < 8; ++r) acc[g][r] = bval[g];
        }
        // gates GEMM: [16 x 640] (LDS, bf16) @ Wcat^T tiles (L2-resident).
#pragma unroll 1
        for (int k = 0; k < K_; k += 32) {
            FragBF af;  // A: sample-major 16x32 tile from LDS
            af.q[0] = *(const u32x4*)&xh[n][k + kh * 16];
            af.q[1] = *(const u32x4*)&xh[n][k + kh * 16 + 8];
            // load ALL four gate B fragments first (distinct regs -> one
            // load clause, decreasing s_wait_loadcnt before each wmma)
            FragBF bfr[4];
#pragma unroll
            for (int g = 0; g < 4; ++g) {
                const unsigned short* wr =
                    wcat + (size_t)(g * H_ + j0 + n) * K_ + k + kh * 16;
                bfr[g].q[0] = *(const u32x4*)wr;
                bfr[g].q[1] = *(const u32x4*)(wr + 8);
            }
            if (k + 32 < K_)   // global_prefetch_b8 of next Wcat K-tile
                __builtin_prefetch(wcat + (size_t)(j0 + n) * K_ + k + 32, 0, 3);
#pragma unroll
            for (int g = 0; g < 4; ++g)
                acc[g] = __builtin_amdgcn_wmma_f32_16x16x32_bf16(
                    false, af.bf, false, bfr[g].bf, (short)0, acc[g], false, false);
        }
        // element-wise LSTM cell (VALU + v_exp_f32/v_rcp_f32 trans ops)
        v8f hnew;
#pragma unroll
        for (int r = 0; r < 8; ++r) {
            float ig = sigm(acc[0][r]);
            float fg = sigm(acc[1][r]);
            float gg = tanh_f(acc[2][r]);
            float og = sigm(acc[3][r]);
            float c  = fg * creg[r] + ig * gg;
            creg[r]  = c;
            hnew[r]  = og * tanh_f(c);
        }
        __syncthreads();   // all waves finished reading xh for this step
#pragma unroll
        for (int r = 0; r < 8; ++r) {
            unsigned short hb = f2bf(hnew[r]);
            int m = r + 8 * kh;
            xh[m][A_ + j0 + n] = hb;                                    // next step
            h_all[((size_t)(bt * 16 + m) * T_ + t) * H_ + j0 + n] = hb; // for k2
        }
        if (t + 1 < T_) {   // stage x_{t+1}
            for (int e = tid; e < 16 * A_; e += 1024) {
                int m = e >> 7, a = e & (A_ - 1);
                xh[m][a] =
                    f2bf(inputs[((size_t)(bt * 16 + m) * T_ + (t + 1)) * A_ + a]);
            }
        }
        __syncthreads();
    }
}

// ---------------------------------------------------------------------------
// Kernel 2: fast-weight augmentation + output projection, fully parallel.
// Block = (sample b, 32-row t-tile); 8 waves / 256 threads; 1024 blocks.
// Stage 1: Haug[32 x 512] = H + H @ M[b]^T   (WMMA, M bf16 from L2)
// Stage 2: Out[32 x 128]  = Haug @ W_out^T + b_out (WMMA, Haug from LDS)
// ---------------------------------------------------------------------------
__global__ void __launch_bounds__(256)
aug_out_kernel(const unsigned short* __restrict__ h_all,
               const unsigned short* __restrict__ mbf,
               const unsigned short* __restrict__ wout,
               const float* __restrict__ b_out,
               float* __restrict__ out) {
    __shared__ unsigned short haug[32][H_];  // 32KB bf16

    const int b    = blockIdx.x >> 4;
    const int t0   = (blockIdx.x & 15) * 32;
    const int tid  = threadIdx.x;
    const int wave = tid >> 5;
    const int lane = tid & 31;
    const int n    = lane & 15;
    const int kh   = lane >> 4;

    // ---- Stage 1: wave = (tq in 0..1, iq in 0..3) -> 16 t-rows x 128 i-cols
    {
        const int tq = wave >> 2;            // t half
        const int iq = wave & 3;             // i group
        const int tb = t0 + tq * 16;
        const int i0 = iq * 128;             // 8 N-tiles of 16
        v8f acc[8];
#pragma unroll
        for (int nt = 0; nt < 8; ++nt) {     // init with the "+ h" identity term
#pragma unroll
            for (int r = 0; r < 8; ++r)
                acc[nt][r] = bf2f(
                    h_all[((size_t)b * T_ + tb + r + 8 * kh) * H_ + i0 + nt * 16 + n]);
        }
#pragma unroll 1
        for (int k = 0; k < H_; k += 32) {
            FragBF af;  // A = H rows (t-major) from h_all
            const unsigned short* hr =
                h_all + ((size_t)b * T_ + tb + n) * H_ + k + kh * 16;
            af.q[0] = *(const u32x4*)hr;
            af.q[1] = *(const u32x4*)(hr + 8);
            // distinct regs for all 8 B fragments -> one big load clause
            FragBF bm[8];
#pragma unroll
            for (int nt = 0; nt < 8; ++nt) { // B = M[b]^T tiles (row i contiguous j)
                const unsigned short* mr =
                    mbf + ((size_t)b * H_ + i0 + nt * 16 + n) * H_ + k + kh * 16;
                bm[nt].q[0] = *(const u32x4*)mr;
                bm[nt].q[1] = *(const u32x4*)(mr + 8);
            }
            if (k + 32 < H_)
                __builtin_prefetch(mbf + ((size_t)b * H_ + i0 + n) * H_ + k + 32, 0, 3);
#pragma unroll
            for (int nt = 0; nt < 8; ++nt)
                acc[nt] = __builtin_amdgcn_wmma_f32_16x16x32_bf16(
                    false, af.bf, false, bm[nt].bf, (short)0, acc[nt], false, false);
        }
#pragma unroll
        for (int nt = 0; nt < 8; ++nt) {
#pragma unroll
            for (int r = 0; r < 8; ++r)
                haug[tq * 16 + r + 8 * kh][i0 + nt * 16 + n] = f2bf(acc[nt][r]);
        }
    }
    __syncthreads();

    // ---- Stage 2: wave = (mt in 0..1, nq in 0..3) -> 16 t-rows x 32 out-cols
    {
        const int mt = wave >> 2;
        const int nq = wave & 3;
        v8f acc[2];
#pragma unroll
        for (int nt = 0; nt < 2; ++nt) {
            float bv = b_out[(nq * 2 + nt) * 16 + n];
#pragma unroll
            for (int r = 0; r < 8; ++r) acc[nt][r] = bv;
        }
#pragma unroll 1
        for (int k = 0; k < H_; k += 32) {
            FragBF af;
            af.q[0] = *(const u32x4*)&haug[mt * 16 + n][k + kh * 16];
            af.q[1] = *(const u32x4*)&haug[mt * 16 + n][k + kh * 16 + 8];
            FragBF bw[2];
#pragma unroll
            for (int nt = 0; nt < 2; ++nt) {
                const unsigned short* wr =
                    wout + (size_t)((nq * 2 + nt) * 16 + n) * H_ + k + kh * 16;
                bw[nt].q[0] = *(const u32x4*)wr;
                bw[nt].q[1] = *(const u32x4*)(wr + 8);
            }
#pragma unroll
            for (int nt = 0; nt < 2; ++nt)
                acc[nt] = __builtin_amdgcn_wmma_f32_16x16x32_bf16(
                    false, af.bf, false, bw[nt].bf, (short)0, acc[nt], false, false);
        }
#pragma unroll
        for (int nt = 0; nt < 2; ++nt) {
#pragma unroll
            for (int r = 0; r < 8; ++r) {
                int t  = t0 + mt * 16 + r + 8 * kh;
                int ac = (nq * 2 + nt) * 16 + n;
                out[((size_t)b * T_ + t) * A_ + ac] = acc[nt][r];
            }
        }
    }
}

extern "C" void kernel_launch(void* const* d_in, const int* in_sizes, int n_in,
                              void* d_out, int out_size, void* d_ws, size_t ws_size,
                              hipStream_t stream) {
    const float* inputs = (const float*)d_in[0];
    const float* w_ih   = (const float*)d_in[1];
    const float* w_hh   = (const float*)d_in[2];
    const float* b_ih   = (const float*)d_in[3];
    const float* b_hh   = (const float*)d_in[4];
    const float* M      = (const float*)d_in[5];
    const float* w_out  = (const float*)d_in[6];
    const float* b_out  = (const float*)d_in[7];
    const float* h0     = (const float*)d_in[8];
    const float* c0     = (const float*)d_in[9];
    float* out = (float*)d_out;

    char* ws = (char*)d_ws;
    unsigned short* wcat  = (unsigned short*)(ws + WCAT_OFF);
    unsigned short* woutb = (unsigned short*)(ws + WOUT_OFF);
    float*          bias  = (float*)(ws + BIAS_OFF);
    unsigned short* mbf   = (unsigned short*)(ws + MBF_OFF);
    unsigned short* hall  = (unsigned short*)(ws + HALL_OFF);

    prep_kernel<<<8192, 256, 0, stream>>>(w_ih, w_hh, b_ih, b_hh, M, w_out,
                                          wcat, woutb, bias, mbf);
    lstm_recurrent_kernel<<<B_ / 16, 1024, 0, stream>>>(inputs, wcat, bias,
                                                        h0, c0, hall);
    aug_out_kernel<<<B_ * (T_ / 32), 256, 0, stream>>>(hall, mbf, woutb,
                                                       b_out, out);
}
